// DeltaNet_22488448762117
// MI455X (gfx1250) — compile-verified, matrix operand order checked by reading, not run on hardware
//
#include <hip/hip_runtime.h>
#include <hip/hip_bf16.h>

// ---------------------------------------------------------------------------
// DeltaNet block for MI455X (gfx1250): bf16 WMMA GEMMs + fp32 recurrent scan
//  - GEMM tiles double-buffered via global_load_async_to_lds_b128 (ASYNCcnt),
//    DMA of tile i+1 overlaps the WMMAs of tile i
//  - single-body K-loop (unroll 1) + offset-XOR buffer toggle so accumulator
//    C/D stay in one register tuple (no v_mov_b64 shuffles)
// ---------------------------------------------------------------------------

#define Bn   4
#define Tn   2048
#define Hn   1024
#define NHn  16
#define Dn   64
#define Mn   (Bn * Tn)        // 8192 tokens
#define EPSf 1e-6f
#define KTILES (Hn / 32)      // 32 K-steps

typedef __attribute__((ext_vector_type(16))) __bf16 v16bf;
typedef __attribute__((ext_vector_type(8)))  float  v8f;

union BFrag {
    v16bf v;
    unsigned short u[16];
    unsigned int   w[8];
    uint4          q[2];
};

static __device__ __forceinline__ unsigned short f2bf(float f) {
    union { float f; unsigned u; } x; x.f = f;
    unsigned r = x.u + 0x7FFFu + ((x.u >> 16) & 1u);   // round-to-nearest-even
    return (unsigned short)(r >> 16);
}

// Low 32 bits of a generic pointer to LDS == LDS byte offset (aperture is in
// addr[63:32] on CDNA5).
static __device__ __forceinline__ unsigned lds_off(const void* p) {
    return (unsigned)(reinterpret_cast<uintptr_t>(p));
}

// Async 16B copy global -> LDS (per-lane), tracked with ASYNCcnt.
static __device__ __forceinline__ void async_copy_b128(unsigned ldsaddr,
                                                       const void* gaddr) {
    asm volatile("global_load_async_to_lds_b128 %0, %1, off"
                 :: "v"(ldsaddr), "v"((unsigned long long)(uintptr_t)gaddr)
                 : "memory");
}
static __device__ __forceinline__ void wait_async0() {
    asm volatile("s_wait_asynccnt 0x0" ::: "memory");
}

// ---------------------------------------------------------------------------
// 1) Weight fp32 -> bf16 convert, layout preserved ([out][in] row-major).
//    GEMM consumes B[k][n] = W[n][k], so original layout is already K-major.
// ---------------------------------------------------------------------------
__global__ void wconv_kernel(const float* __restrict__ Wq, const float* __restrict__ Wk,
                             const float* __restrict__ Wv, const float* __restrict__ Wg,
                             const float* __restrict__ Wo, unsigned short* __restrict__ WbAll) {
    const float* W;
    switch (blockIdx.y) {
        case 0: W = Wq; break;
        case 1: W = Wk; break;
        case 2: W = Wv; break;
        case 3: W = Wg; break;
        default: W = Wo; break;
    }
    unsigned short* O = WbAll + (size_t)blockIdx.y * Hn * Hn;
    const size_t i = ((size_t)blockIdx.x * 256 + threadIdx.x) * 4;
    const float4 f = *(const float4*)(W + i);
    uint2 o;
    o.x = (unsigned)f2bf(f.x) | ((unsigned)f2bf(f.y) << 16);
    o.y = (unsigned)f2bf(f.z) | ((unsigned)f2bf(f.w) << 16);
    *(uint2*)(O + i) = o;
}

// ---------------------------------------------------------------------------
// 2) RMSNorm over H=1024 per token, write bf16 xn. 256 thr/token, 4 cols each.
// ---------------------------------------------------------------------------
__global__ void rmsnorm_kernel(const float* __restrict__ x, const float* __restrict__ w,
                               unsigned short* __restrict__ xn) {
    const int row = blockIdx.x, tid = threadIdx.x;
    const float* xr = x + (size_t)row * Hn;
    float vals[4];
    float s = 0.f;
#pragma unroll
    for (int i = 0; i < 4; ++i) {
        float v = xr[tid + i * 256];
        vals[i] = v;
        s += v * v;
    }
    __shared__ float red[256];
    red[tid] = s;
    __syncthreads();
    for (int st = 128; st > 0; st >>= 1) {
        if (tid < st) red[tid] += red[tid + st];
        __syncthreads();
    }
    const float rms = rsqrtf(red[0] * (1.0f / Hn) + EPSf);
#pragma unroll
    for (int i = 0; i < 4; ++i) {
        const int c = tid + i * 256;
        xn[(size_t)row * Hn + c] = f2bf(vals[i] * rms * w[c]);
    }
}

// ---------------------------------------------------------------------------
// 3) bf16 WMMA GEMM: Out[m,n] = sum_k A[m,k] * W[n,k]   (N=K=1024)
//    Block tile 128(M) x 64(N); 8 waves (wave32), each wave -> 32x32
//    = 4 x v_wmma_f32_16x16x32_bf16 per K-step from 8 ds_load_b128.
//    mode 0: raw fp32 store, weights/out selected by blockIdx.z (q/k/v/g).
//    mode 1: Out = acc + bias[n] + resid[m,n]  (output proj + residual).
// ---------------------------------------------------------------------------
__global__ void gemm_bf16_kernel(const unsigned short* __restrict__ A,
                                 const unsigned short* __restrict__ WbBase,
                                 float* __restrict__ OutBase,
                                 const float* __restrict__ bias,
                                 const float* __restrict__ resid,
                                 int Mrows, int mode) {
    const int bm = blockIdx.x * 128;
    const int bn = blockIdx.y * 64;
    const unsigned short* Wb = WbBase + (size_t)blockIdx.z * Hn * Hn;
    float* Out = OutBase + (mode == 0 ? (size_t)blockIdx.z * (size_t)Mrows * Hn : (size_t)0);

    __shared__ unsigned short lA[2 * 128 * 32];   // 2 x 8 KB, row = M, K contiguous
    __shared__ unsigned short lB[2 * 64 * 32];    // 2 x 4 KB, row = N, K contiguous

    const int tid  = threadIdx.x;
    const int lane = tid & 31;
    const int wave = tid >> 5;          // 0..7
    const int wm   = wave & 3;          // M sub-tile (x32)
    const int wn   = wave >> 2;         // N sub-tile (x32)
    const int fm   = lane & 15;         // fragment row/col within 16
    const int fhi  = lane >> 4;         // half-wave -> K+8 striping

    v8f acc00 = {0.f,0.f,0.f,0.f,0.f,0.f,0.f,0.f};
    v8f acc01 = acc00, acc10 = acc00, acc11 = acc00;

    const int sRow = tid >> 2;          // 0..63
    const int sCh  = tid & 3;           // 4 chunks of 8 bf16 (16B)

    // issue ASYNCcnt-tracked DMA of one K-tile into buffer at element offsets
    auto stage = [&](unsigned oA, unsigned oB, int k0) {
        async_copy_b128(lds_off(&lA[oA + sRow * 32 + sCh * 8]),
                        A + (size_t)(bm + sRow) * Hn + k0 + sCh * 8);
        async_copy_b128(lds_off(&lA[oA + (sRow + 64) * 32 + sCh * 8]),
                        A + (size_t)(bm + sRow + 64) * Hn + k0 + sCh * 8);
        async_copy_b128(lds_off(&lB[oB + sRow * 32 + sCh * 8]),
                        Wb + (size_t)(bn + sRow) * Hn + k0 + sCh * 8);
    };

    // 4 WMMAs from the buffer at element offsets (2x ds_load_b128 per frag)
    auto compute = [&](unsigned oA, unsigned oB) {
        BFrag a0, a1, b0, b1;
        const unsigned short* pa0 = &lA[oA + (wm * 32 + fm) * 32];
        const unsigned short* pa1 = &lA[oA + (wm * 32 + 16 + fm) * 32];
        const unsigned short* pb0 = &lB[oB + (wn * 32 + fm) * 32];
        const unsigned short* pb1 = &lB[oB + (wn * 32 + 16 + fm) * 32];
        a0.q[0] = *(const uint4*)(pa0 + fhi * 8);
        a0.q[1] = *(const uint4*)(pa0 + 16 + fhi * 8);
        a1.q[0] = *(const uint4*)(pa1 + fhi * 8);
        a1.q[1] = *(const uint4*)(pa1 + 16 + fhi * 8);
        b0.q[0] = *(const uint4*)(pb0 + fhi * 8);
        b0.q[1] = *(const uint4*)(pb0 + 16 + fhi * 8);
        b1.q[0] = *(const uint4*)(pb1 + fhi * 8);
        b1.q[1] = *(const uint4*)(pb1 + 16 + fhi * 8);
        acc00 = __builtin_amdgcn_wmma_f32_16x16x32_bf16(false, a0.v, false, b0.v,
                                                        (short)0, acc00, false, false);
        acc01 = __builtin_amdgcn_wmma_f32_16x16x32_bf16(false, a0.v, false, b1.v,
                                                        (short)0, acc01, false, false);
        acc10 = __builtin_amdgcn_wmma_f32_16x16x32_bf16(false, a1.v, false, b0.v,
                                                        (short)0, acc10, false, false);
        acc11 = __builtin_amdgcn_wmma_f32_16x16x32_bf16(false, a1.v, false, b1.v,
                                                        (short)0, acc11, false, false);
    };

    unsigned offA = 0, offB = 0;        // element offsets of current buffer
    stage(0, 0, 0);                     // prologue DMA

#pragma unroll 1
    for (int it = 0; it < KTILES - 1; ++it) {
        wait_async0();                  // tile `it` landed in current buffer
        __syncthreads();                // ... for every wave; prev readers done
        stage(offA ^ 4096u, offB ^ 2048u, (it + 1) * 32);   // overlaps WMMAs
        __builtin_prefetch(A  + (size_t)(bm + sRow) * Hn + (it + 2) * 32 + sCh * 8, 0, 1);
        __builtin_prefetch(Wb + (size_t)(bn + sRow) * Hn + (it + 2) * 32 + sCh * 8, 0, 1);
        compute(offA, offB);
        offA ^= 4096u;
        offB ^= 2048u;
    }
    wait_async0();                      // last tile
    __syncthreads();
    compute(offA, offB);

    // C layout: VGPR r -> M = r (lanes 0-15) / r+8 (lanes 16-31); N = lane%16
#pragma unroll
    for (int mi = 0; mi < 2; ++mi) {
        const v8f* accs0 = (mi == 0) ? &acc00 : &acc10;
        const v8f* accs1 = (mi == 0) ? &acc01 : &acc11;
#pragma unroll
        for (int r = 0; r < 8; ++r) {
            const int gm  = bm + wm * 32 + mi * 16 + r + fhi * 8;
            const int gc0 = bn + wn * 32 + fm;
            const int gc1 = gc0 + 16;
            const size_t i0 = (size_t)gm * Hn + gc0;
            const size_t i1 = (size_t)gm * Hn + gc1;
            if (mode == 0) {
                Out[i0] = (*accs0)[r];
                Out[i1] = (*accs1)[r];
            } else {
                Out[i0] = (*accs0)[r] + bias[gc0] + resid[i0];
                Out[i1] = (*accs1)[r] + bias[gc1] + resid[i1];
            }
        }
    }
}

// ---------------------------------------------------------------------------
// 4) RoPE + elu-feature + sigmoid gate + mask (pair-safe, in-place).
//    cos[i] = cos(t * 10000^(-(i/2)/32)); rotate_half mixes d and d±32.
// ---------------------------------------------------------------------------
__global__ void features_kernel(float* __restrict__ q, float* __restrict__ k,
                                float* __restrict__ v, float* __restrict__ g,
                                const int* __restrict__ mask, const float* __restrict__ bg) {
    const int bt = blockIdx.x;
    const int t  = bt % Tn;
    const float m = (float)mask[bt];
    const size_t base = (size_t)bt * Hn;
    const float LOG_BASE_OVER_HALF = 0.28782313662425572f;  // ln(10000)/32

    for (int p = threadIdx.x; p < Hn / 2; p += 256) {
        const int h  = p >> 5;          // head
        const int i  = p & 31;          // d in [0,32)
        const int c1 = h * 64 + i;      // partner pair (c1, c1+32)
        const int c2 = c1 + 32;

        const float q1 = q[base + c1], q2 = q[base + c2];
        const float k1 = k[base + c1], k2 = k[base + c2];

        const int j1 = i >> 1, j2 = (i + 32) >> 1;
        const float a1 = (float)t * __expf(-LOG_BASE_OVER_HALF * (float)j1);
        const float a2 = (float)t * __expf(-LOG_BASE_OVER_HALF * (float)j2);
        const float cA = cosf(a1), sA = sinf(a1);
        const float cB = cosf(a2), sB = sinf(a2);

        const float qr1 = q1 * cA - q2 * sA;   // rotate_half: -x2 for first half
        const float qr2 = q2 * cB + q1 * sB;   //              +x1 for second half
        const float kr1 = k1 * cA - k2 * sA;
        const float kr2 = k2 * cB + k1 * sB;

        // phi = elu(x) + 1  ->  x>0 ? x+1 : exp(x)
        q[base + c1] = (qr1 > 0.f) ? qr1 + 1.f : __expf(qr1);
        q[base + c2] = (qr2 > 0.f) ? qr2 + 1.f : __expf(qr2);
        k[base + c1] = m * ((kr1 > 0.f) ? kr1 + 1.f : __expf(kr1));
        k[base + c2] = m * ((kr2 > 0.f) ? kr2 + 1.f : __expf(kr2));

        v[base + c1] *= m;
        v[base + c2] *= m;

        const float g1 = g[base + c1] + bg[c1];
        const float g2 = g[base + c2] + bg[c2];
        const float r1 = 1.f / (1.f + __expf(-g1));
        const float r2 = 1.f / (1.f + __expf(-g2));
        g[base + c1] = (m > 0.f) ? r1 : 1.f;
        g[base + c2] = (m > 0.f) ? r2 : 1.f;
    }
}

// ---------------------------------------------------------------------------
// 5) Recurrent scan: one block per (b,h); 64 lanes, lane e owns S[:,e] in
//    64 registers + Z[e]. Per step one float4 LDS broadcast per d.
//    S[d][e] = S[d][e]*r[d] + k[d]*v[e];  num[e]=sum_d q[d]S[d][e];
//    Z[d]=Z[d]*r[d]+k[d];  den=sum_d q[d]Z[d];  y=num/(den+eps) -> bf16
// ---------------------------------------------------------------------------
__global__ void scan_kernel(const float* __restrict__ q, const float* __restrict__ k,
                            const float* __restrict__ v, const float* __restrict__ r,
                            unsigned short* __restrict__ y) {
    const int blk = blockIdx.x;
    const int b = blk >> 4, h = blk & 15;
    const int e = threadIdx.x;   // 0..63

    __shared__ float4 sh[64];    // {q[d], k[d], r[d], q[d]*Z[d]}

    float S[64];
#pragma unroll
    for (int d = 0; d < 64; ++d) S[d] = 0.f;
    float z = 0.f;               // Z[e]

    for (int t = 0; t < Tn; ++t) {
        const size_t base = ((size_t)(b * Tn + t)) * Hn + h * 64;
        const float qe = q[base + e];
        const float ke = k[base + e];
        const float ve = v[base + e];
        const float re = r[base + e];

        z = z * re + ke;
        sh[e] = make_float4(qe, ke, re, qe * z);
        __syncthreads();

        float num = 0.f, den = 0.f;
#pragma unroll
        for (int d = 0; d < 64; ++d) {
            const float4 w = sh[d];           // single ds_load_b128 broadcast
            S[d] = S[d] * w.z + w.y * ve;     // decay + outer-product column
            num  = fmaf(w.x, S[d], num);
            den += w.w;
        }
        y[base + e] = f2bf(num / (den + EPSf));
        __syncthreads();
    }
}

// ---------------------------------------------------------------------------
// Launch
// ---------------------------------------------------------------------------
extern "C" void kernel_launch(void* const* d_in, const int* in_sizes, int n_in,
                              void* d_out, int out_size, void* d_ws, size_t ws_size,
                              hipStream_t stream) {
    (void)in_sizes; (void)n_in; (void)out_size; (void)ws_size;

    const float* x     = (const float*)d_in[0];
    const int*   mask  = (const int*)  d_in[1];
    const float* Wq    = (const float*)d_in[2];
    const float* Wk    = (const float*)d_in[3];
    const float* Wv    = (const float*)d_in[4];
    const float* Wg    = (const float*)d_in[5];
    const float* Wo    = (const float*)d_in[6];
    const float* bo    = (const float*)d_in[7];
    const float* bg    = (const float*)d_in[8];
    const float* rms_w = (const float*)d_in[9];

    // workspace carve-up
    char* ws = (char*)d_ws;
    size_t off = 0;
    unsigned short* Wb = (unsigned short*)(ws + off); off += (size_t)5 * Hn * Hn * 2;   // 10 MB
    unsigned short* xn = (unsigned short*)(ws + off); off += (size_t)Mn * Hn * 2;       // 16 MB
    float* qkvg        = (float*)(ws + off);          off += (size_t)4 * Mn * Hn * 4;   // 128 MB
    unsigned short* ya = (unsigned short*)(ws + off); off += (size_t)Mn * Hn * 2;       // 16 MB

    float* qb = qkvg;
    float* kb = qkvg + (size_t)1 * Mn * Hn;
    float* vb = qkvg + (size_t)2 * Mn * Hn;
    float* gb = qkvg + (size_t)3 * Mn * Hn;

    // 1) weights -> bf16 (layout preserved; already B-operand friendly)
    wconv_kernel<<<dim3((Hn * Hn) / 1024, 5), 256, 0, stream>>>(
        Wq, Wk, Wv, Wg, Wo, Wb);

    // 2) RMSNorm -> bf16 xn
    rmsnorm_kernel<<<Mn, 256, 0, stream>>>(x, rms_w, xn);

    // 3) fused q/k/v/g projection GEMMs (WMMA bf16, pipelined async staging)
    gemm_bf16_kernel<<<dim3(Mn / 128, Hn / 64, 4), 256, 0, stream>>>(
        xn, Wb, qkvg, nullptr, nullptr, Mn, 0);

    // 4) RoPE + feature map + gate + mask (in-place)
    features_kernel<<<Mn, 256, 0, stream>>>(qb, kb, vb, gb, mask, bg);

    // 5) recurrent scan -> bf16 attention output
    scan_kernel<<<Bn * NHn, 64, 0, stream>>>(qb, kb, vb, gb, ya);

    // 6) output projection + bias + residual -> d_out (fp32)
    gemm_bf16_kernel<<<dim3(Mn / 128, Hn / 64, 1), 256, 0, stream>>>(
        ya, Wb + (size_t)4 * Hn * Hn, (float*)d_out, bo, x, Mn, 1);
}
